// Encoder_40535901340423
// MI455X (gfx1250) — matrix operand order repaired and checked by reading, not compiled, not run on hardware
//
#include <hip/hip_runtime.h>
#include <hip/hip_bf16.h>
#include <cstdint>
#include <cstddef>

typedef _Float16 v16h __attribute__((ext_vector_type(16)));
typedef _Float16 v8h  __attribute__((ext_vector_type(8)));
typedef float    v8f  __attribute__((ext_vector_type(8)));

#define HID  128
#define VOC  32
#define KTOT 288      // 32 (x) + 128 (h1) + 128 (h2)
#define NCOL 512      // 4 gate groups (r, z, n_x, n_h) x 128
#define ROWS_WG 64    // 64 rows/WG; B=128 => node index j constant per WG
#define H_STRIDE 264  // halfs per LDS h row (528B = 33*16)
#define PRE_STRIDE 512

// ---------------------------------------------------------------------------
// Pack the six GRU weight matrices into one f16 [NCOL][KTOT] B-operand matrix
// (row n = output column, contiguous K) + combined bias[NCOL].
//   cols   0..127 : r gate   K: [Wir | Whr]
//   cols 128..255 : z gate   K: [Wiz | Whz]
//   cols 256..383 : n x-part K: [Win | 0  ]
//   cols 384..511 : n h-part K: [ 0  | Whn]
// ---------------------------------------------------------------------------
__global__ __launch_bounds__(256)
void prep_weights(const float* __restrict__ wir, const float* __restrict__ whr,
                  const float* __restrict__ wiz, const float* __restrict__ whz,
                  const float* __restrict__ win, const float* __restrict__ whn,
                  const float* __restrict__ bir, const float* __restrict__ bhr,
                  const float* __restrict__ biz, const float* __restrict__ bhz,
                  const float* __restrict__ bin, const float* __restrict__ bhn,
                  _Float16* __restrict__ Wt, float* __restrict__ bias) {
  int idx = blockIdx.x * blockDim.x + threadIdx.x;
  if (idx < NCOL) {
    int g = idx >> 7, o = idx & 127;
    float bv = (g == 0) ? bir[o] + bhr[o]
             : (g == 1) ? biz[o] + bhz[o]
             : (g == 2) ? bin[o] : bhn[o];
    bias[idx] = bv;
  }
  const int total = NCOL * KTOT;
  for (int e = idx; e < total; e += gridDim.x * blockDim.x) {
    int n = e / KTOT, k = e - n * KTOT;
    int g = n >> 7, o = n & 127;
    float v;
    if      (g == 0) v = (k < VOC) ? wir[o * VOC + k] : whr[o * 256 + (k - VOC)];
    else if (g == 1) v = (k < VOC) ? wiz[o * VOC + k] : whz[o * 256 + (k - VOC)];
    else if (g == 2) v = (k < VOC) ? win[o * VOC + k] : 0.f;
    else             v = (k < VOC) ? 0.f               : whn[o * 256 + (k - VOC)];
    Wt[(size_t)n * KTOT + k] = (_Float16)v;
  }
}

// Pre-convert all targets to f16 once (enables pure byte-copy staging).
__global__ __launch_bounds__(256)
void conv_targets(const float* __restrict__ t, _Float16* __restrict__ t16, int total4) {
  int idx = blockIdx.x * blockDim.x + threadIdx.x;
  for (int i = idx; i < total4; i += gridDim.x * blockDim.x) {
    float4 v = ((const float4*)t)[i];
    _Float16* o = t16 + (size_t)i * 4;
    o[0] = (_Float16)v.x; o[1] = (_Float16)v.y;
    o[2] = (_Float16)v.z; o[3] = (_Float16)v.w;
  }
}

// issue one 16B global->LDS async copy (GVS mode: saddr + 32-bit voffset)
__device__ __forceinline__
void async_copy16(uint32_t lds_byte_off, uint32_t gmem_byte_off, const void* sbase) {
  asm volatile("global_load_async_to_lds_b128 %0, %1, %2"
               :: "v"(lds_byte_off), "v"(gmem_byte_off), "s"(sbase) : "memory");
}

// ---------------------------------------------------------------------------
// One tree level: rows = n*B GRU221 cells.  64 rows x 512 pre-act cols per WG.
// Wave w: gate group cg = w&3, col half ch = w>>2 (4 col tiles), all 4 row
// tiles -> each B fragment loaded once, used 4x.  f16 WMMA, f32 accumulate.
// ---------------------------------------------------------------------------
__global__ __launch_bounds__(256)
void tree_gru_level(const _Float16* __restrict__ targetsL,  // t16 + (n-1)*B*V
                    const float* __restrict__ masksL,       // masks + (n-1)*B
                    const _Float16* __restrict__ hprev,     // level d+1 h (f16), null at leaves
                    _Float16* __restrict__ hcur,            // level d h (f16)
                    const _Float16* __restrict__ Wt,
                    const float* __restrict__ bias,
                    int has_children) {
  __shared__ __align__(16) union {
    struct {
      _Float16 x[ROWS_WG][VOC];       //  4 KB, row = 64B
      _Float16 h[ROWS_WG][H_STRIDE];  // 33 KB, row = 528B: [h1(256B) | h2(256B) | pad]
    } in;
    float pre[ROWS_WG][PRE_STRIDE];   // 128 KB f32 pre-activations (after barrier)
  } sm;

  const int tid  = threadIdx.x;
  const int row0 = blockIdx.x * ROWS_WG;
  const int jnode = row0 >> 7;        // constant across WG (64 | 128)
  const int b0    = row0 & 127;

  // ---- phase 1: async-stage x / h1 / h2 blocks (all contiguous in global) --
  {
    const uint32_t xlds = (uint32_t)(uintptr_t)&sm.in.x[0][0];
    const void* xg = (const void*)(targetsL + (size_t)row0 * VOC);
    async_copy16(xlds + tid * 16, tid * 16, xg);  // 64*32*2 = 4096B, 1 per thread

    if (has_children) {
      const uint32_t hlds = (uint32_t)(uintptr_t)&sm.in.h[0][0];
      const void* h1g = (const void*)(hprev + ((size_t)(2 * jnode)     * 128 + b0) * HID);
      const void* h2g = (const void*)(hprev + ((size_t)(2 * jnode + 1) * 128 + b0) * HID);
      #pragma unroll
      for (int it = 0; it < 4; ++it) {            // 64 rows * 256B each = 1024 chunks
        int c = tid + it * 256;
        uint32_t r = (uint32_t)c >> 4;
        uint32_t off = ((uint32_t)c & 15) * 16;
        uint32_t dst = hlds + r * (H_STRIDE * 2) + off;
        async_copy16(dst,       c * 16, h1g);
        async_copy16(dst + 256, c * 16, h2g);
      }
    } else {
      uint4* hz = (uint4*)&sm.in.h[0][0];         // zero children (leaves)
      for (int e = tid; e < ROWS_WG * H_STRIDE * 2 / 16; e += 256)
        hz[e] = make_uint4(0u, 0u, 0u, 0u);
    }
    asm volatile("s_wait_asynccnt 0" ::: "memory");
  }
  __syncthreads();

  // ---- phase 2: WMMA GEMM, acc = A[64 x 288] * Wt^T ----
  const int lane = tid & 31;
  const int wave = tid >> 5;
  const int cg   = wave & 3;    // gate group
  const int ch   = wave >> 2;   // column half within group
  const int m    = lane & 15;   // A row / B col within tile
  const int hi   = lane >> 4;

  v8f acc[4][4] = {};           // [row tile][col tile]

  const int kc_lo = (cg == 3) ? 1 : 0;     // n_h skips x chunk
  const int kc_hi = (cg == 2) ? 1 : 9;     // n_x is x chunk only
  for (int kc = kc_lo; kc < kc_hi; ++kc) {
    v16h Bf[4];
    #pragma unroll
    for (int u = 0; u < 4; ++u) {
      const _Float16* wp = Wt + (size_t)(cg * 128 + (ch * 4 + u) * 16 + m) * KTOT
                              + kc * 32 + hi * 16;
      Bf[u] = *(const v16h*)wp;            // 32B contiguous, 32B aligned
    }
    #pragma unroll
    for (int rt = 0; rt < 4; ++rt) {
      union { v16h v; v8h h[2]; } A;
      const _Float16* ap = (kc == 0)
          ? &sm.in.x[rt * 16 + m][hi * 16]
          : &sm.in.h[rt * 16 + m][(kc - 1) * 32 + hi * 16];
      A.h[0] = *(const v8h*)ap;            // K = base..base+7
      A.h[1] = *(const v8h*)(ap + 8);      // K = base+8..base+15
      #pragma unroll
      for (int u = 0; u < 4; ++u)
        acc[rt][u] = __builtin_amdgcn_wmma_f32_16x16x32_f16(
            false, A.v, false, Bf[u], (short)0, acc[rt][u], false, false);
    }
  }
  __syncthreads();  // all waves done reading sm.in

  // ---- phase 3: spill pre-activations (C/D layout: VGPR j -> M = hi*8+j) ---
  #pragma unroll
  for (int rt = 0; rt < 4; ++rt) {
    int mrow = rt * 16 + hi * 8;
    #pragma unroll
    for (int u = 0; u < 4; ++u) {
      int col = cg * 128 + (ch * 4 + u) * 16 + m;
      #pragma unroll
      for (int j = 0; j < 8; ++j)
        sm.pre[mrow + j][col] = acc[rt][u][j];
    }
  }
  __syncthreads();

  // ---- phase 4: fp32 gate epilogue, f16 state out ----
  for (int e = tid; e < ROWS_WG * HID; e += 256) {
    int r = e >> 7, o = e & 127;
    int gr = row0 + r;
    float rv = 1.f / (1.f + expf(-(sm.pre[r][o]       + bias[o])));
    float zv = 1.f / (1.f + expf(-(sm.pre[r][128 + o] + bias[128 + o])));
    float nh = sm.pre[r][384 + o] + bias[384 + o];
    float cv = tanhf(sm.pre[r][256 + o] + bias[256 + o] + rv * nh);
    float hch = 0.f;
    if (has_children) {
      int b = gr & 127;
      float h1 = (float)hprev[((size_t)(2 * jnode)     * 128 + b) * HID + o];
      float h2 = (float)hprev[((size_t)(2 * jnode + 1) * 128 + b) * HID + o];
      hch = h1 + h2;
    }
    float hout = (1.f - zv) * cv + zv * 0.5f * hch;
    hout *= masksL[gr];
    hcur[(size_t)gr * HID + o] = (_Float16)hout;
  }
}

// ---------------------------------------------------------------------------
// Head: mu = root @ mu_w^T + mu_b ; logvar = root @ lv_w^T + lv_b  (4.2 MFLOP)
// ---------------------------------------------------------------------------
__global__ __launch_bounds__(256)
void head_proj(const _Float16* __restrict__ root,  // [128][128] f16
               const float* __restrict__ mu_w, const float* __restrict__ mu_b,
               const float* __restrict__ lv_w, const float* __restrict__ lv_b,
               float* __restrict__ out) {          // [mu | logvar], 32768 f32
  int idx = blockIdx.x * blockDim.x + threadIdx.x;
  int which = idx >> 14;
  int rem = idx & 16383;
  int b = rem >> 7, o = rem & 127;
  const float* w  = which ? lv_w : mu_w;
  const float* bs = which ? lv_b : mu_b;
  float s = bs[o];
  const _Float16* hr = root + b * HID;
  const float* wr = w + o * HID;
  #pragma unroll 8
  for (int h = 0; h < HID; ++h) s += (float)hr[h] * wr[h];
  out[idx] = s;
}

// ---------------------------------------------------------------------------
extern "C" void kernel_launch(void* const* d_in, const int* in_sizes, int n_in,
                              void* d_out, int out_size, void* d_ws, size_t ws_size,
                              hipStream_t stream) {
  const float* targets = (const float*)d_in[0];
  const float* masks   = (const float*)d_in[1];
  const float* wir_w = (const float*)d_in[2];  const float* wir_b = (const float*)d_in[3];
  const float* whr_w = (const float*)d_in[4];  const float* whr_b = (const float*)d_in[5];
  const float* wiz_w = (const float*)d_in[6];  const float* wiz_b = (const float*)d_in[7];
  const float* whz_w = (const float*)d_in[8];  const float* whz_b = (const float*)d_in[9];
  const float* win_w = (const float*)d_in[10]; const float* win_b = (const float*)d_in[11];
  const float* whn_w = (const float*)d_in[12]; const float* whn_b = (const float*)d_in[13];
  const float* mu_w  = (const float*)d_in[14]; const float* mu_b  = (const float*)d_in[15];
  const float* lv_w  = (const float*)d_in[16]; const float* lv_b  = (const float*)d_in[17];

  // workspace layout
  char* ws = (char*)d_ws;
  _Float16* Wt   = (_Float16*)ws;                       // 512*288*2      = 294,912 B
  float*    bias = (float*)(ws + 294912);               // 512*4          =   2,048 B
  _Float16* t16  = (_Float16*)(ws + 296960);            // 2047*128*32*2  = 16,769,024 B
  _Float16* bufA = (_Float16*)(ws + 17070080);          // 1024*128*128*2 = 33,554,432 B
  _Float16* bufB = (_Float16*)(ws + 50624512);          //  512*128*128*2 = 16,777,216 B

  prep_weights<<<576, 256, 0, stream>>>(wir_w, whr_w, wiz_w, whz_w, win_w, whn_w,
                                        wir_b, whr_b, wiz_b, whz_b, win_b, whn_b,
                                        Wt, bias);
  conv_targets<<<2048, 256, 0, stream>>>(targets, t16, 2047 * 128 * 32 / 4);

  // bottom-up level sweep: d = 10 (1024 leaf-nodes) .. 0 (root)
  const _Float16* hprev = nullptr;
  _Float16* hcur = nullptr;
  for (int d = 10; d >= 0; --d) {
    int n = 1 << d;
    hcur = (d & 1) ? bufB : bufA;     // even levels (incl. 10 and 0) -> bufA
    size_t off = (size_t)(n - 1);     // heap: level starts at node 2^d - 1
    int blocks = (n * 128) / ROWS_WG; // 64 rows per WG
    tree_gru_level<<<blocks, 256, 0, stream>>>(
        t16 + off * 128 * VOC, masks + off * 128,
        hprev, hcur, Wt, bias, hprev != nullptr ? 1 : 0);
    hprev = hcur;
  }

  head_proj<<<128, 256, 0, stream>>>(hcur, mu_w, mu_b, lv_w, lv_b, (float*)d_out);
}
